// ExperimentalModel_1752346656819
// MI455X (gfx1250) — matrix operand order repaired and checked by reading, not compiled, not run on hardware
//
#include <hip/hip_runtime.h>
#include <hip/hip_bf16.h>

// CDNA5 / gfx1250, wave32. WMMA f32 = f16xf16 16x16x32 (codegen-verified builtin).
typedef __attribute__((ext_vector_type(16))) _Float16 v16h;
typedef __attribute__((ext_vector_type(8)))  float    v8f;

#define ASYMF 1e-3f
#define EPSF  0.01f
#define EF    2.71828182845904523f

// ---------------------------------------------------------------------------
// Gather ctx rows and initial latents per sample.
// ---------------------------------------------------------------------------
__global__ __launch_bounds__(256)
void gather_init(const int* __restrict__ ids, const float* __restrict__ ctx_cache,
                 const float* __restrict__ latents, float* __restrict__ ctx,
                 float* __restrict__ x0)
{
    int b  = blockIdx.x;
    int id = ids[b];
    for (int j = threadIdx.x; j < 32; j += blockDim.x)
        ctx[b * 32 + j] = ctx_cache[id * 32 + j];
    for (int j = threadIdx.x; j < 8 * 16 * 16; j += blockDim.x)
        x0[(long)b * 2048 + j] = latents[(long)id * 2048 + j];
}

// ---------------------------------------------------------------------------
// Per-sample dynamic weight generation (templated so all / and % fold):
//   u = ctx@U [R=16,O], v = ctx@V [R,KT], w = W * (1 + u^T v + ASYM), bias = ctx@Bw
// ---------------------------------------------------------------------------
template<int O, int KT>
__global__ __launch_bounds__(256)
void dyn_weight(const float* __restrict__ ctx, const float* __restrict__ W,
                const float* __restrict__ U, const float* __restrict__ V,
                const float* __restrict__ Bw, float* __restrict__ wout,
                float* __restrict__ bout)
{
    __shared__ float ctx_s[32];
    __shared__ float u_s[16 * O];
    __shared__ float v_s[16 * KT];
    int b = blockIdx.x, tid = threadIdx.x;
    if (tid < 32) ctx_s[tid] = ctx[b * 32 + tid];
    __syncthreads();

    constexpr int RO = 16 * O;
    for (int j = tid; j < RO; j += 256) {
        float s = 0.f;
        #pragma unroll
        for (int l = 0; l < 32; ++l) s += ctx_s[l] * U[l * RO + j];
        u_s[j] = s;
    }
    constexpr int RK = 16 * KT;
    for (int j = tid; j < RK; j += 256) {
        float s = 0.f;
        #pragma unroll
        for (int l = 0; l < 32; ++l) s += ctx_s[l] * V[l * RK + j];
        v_s[j] = s;
    }
    __syncthreads();

    constexpr int OK = O * KT;
    for (int j = tid; j < OK; j += 256) {
        int o = j / KT, k = j % KT;
        float mod = 0.f;
        #pragma unroll
        for (int r = 0; r < 16; ++r) mod += u_s[r * O + o] * v_s[r * KT + k];
        wout[(long)b * OK + j] = W[j] * (1.0f + mod + ASYMF);
    }
    if (tid < O) {
        float s = 0.f;
        #pragma unroll
        for (int l = 0; l < 32; ++l) s += ctx_s[l] * Bw[l * O + tid];
        bout[b * O + tid] = s;
    }
}

// ---------------------------------------------------------------------------
// Pack per-sample weights into WMMA A-fragment order (ISA 7.12.2, f16,
// ZERO-padded for m>=8 / k>=KT -> the conv's B tail needs no masking;
// MODE 1 gets the transpose-conv kernel flip baked in).
// afrag[(b*NSTEP + s)*32 + lane][0..15] halves == v16h operand.
// ---------------------------------------------------------------------------
template<int MODE>
__global__ __launch_bounds__(256)
void pack_afrag(const float* __restrict__ w, _Float16* __restrict__ afrag)
{
    constexpr int KT    = (MODE == 0) ? 72 : (MODE == 1) ? 128 : 200;
    constexpr int NSTEP = (MODE == 0) ? 3  : (MODE == 1) ? 4   : 7;
    int tid = blockIdx.x * 256 + threadIdx.x;
    if (tid >= 64 * NSTEP * 32) return;
    int lane = tid & 31;
    int s    = (tid >> 5) % NSTEP;
    int b    = (tid >> 5) / NSTEP;
    int m = lane & 15, kgrp = lane >> 4;
    const float* wb  = w + (long)b * 8 * KT;
    _Float16*    out = afrag + (long)tid * 16;
    int kbase = s * 32;
    #pragma unroll
    for (int j = 0; j < 8; ++j) {
        int kb = ((j < 4) ? 2 * j : 16 + 2 * (j - 4)) + (kgrp ? 8 : 0);
        #pragma unroll
        for (int e = 0; e < 2; ++e) {
            int k = kbase + kb + e;
            float val = 0.f;
            if (m < 8 && k < KT) {
                if (MODE == 1) { // flipped 4x4 kernel for transpose conv
                    int ic = k >> 4, r2 = k & 15, ky = r2 >> 2, kx = r2 & 3;
                    val = wb[m * KT + ic * 16 + (3 - ky) * 4 + (3 - kx)];
                } else {
                    val = wb[m * KT + k];
                }
            }
            out[2 * j + e] = (_Float16)val;
        }
    }
}

// ---------------------------------------------------------------------------
// Init per-map |x| minima to +inf (non-negative float bits order as uints).
// ---------------------------------------------------------------------------
__global__ __launch_bounds__(256)
void init_min(unsigned int* __restrict__ mins)
{
    int i = blockIdx.x * blockDim.x + threadIdx.x;
    if (i < 1536) mins[i] = 0x7F800000u;
}

// ---------------------------------------------------------------------------
// Zero-halo padded f16 copies of the conv inputs (one-time f32->f16 convert,
// so the conv gather is a bare 2-byte load with no masking/clamping/cvt).
// ---------------------------------------------------------------------------
// 3x3 path: halo 1, (H+2)x(W+2) per map.
template<int LOGW>
__global__ __launch_bounds__(256)
void pad_copy_f16(const float* __restrict__ x, _Float16* __restrict__ xp)
{
    constexpr int H = 1 << LOGW, W = H, PH = H + 2, PW = W + 2;
    int idx = blockIdx.x * 256 + threadIdx.x;
    if (idx >= 512 * PH * PW) return;
    int px = idx % PW; int t = idx / PW;
    int py = t % PH;   int bc = t / PH;
    float v = 0.f;
    int yy = py - 1, xx = px - 1;
    if (yy >= 0 && yy < H && xx >= 0 && xx < W)
        v = x[((long)bc << (2 * LOGW)) + (yy << LOGW) + xx];
    xp[idx] = (_Float16)v;
}

// Transpose-conv path: materialize the lhs-dilated (stride-2), negated input
// with halo 2: (2H+4)x(2W+4) per map; dilated grid is (2H-1)x(2W-1).
template<int LOGW>
__global__ __launch_bounds__(256)
void dilate_neg_f16(const float* __restrict__ x, _Float16* __restrict__ dp)
{
    constexpr int H = 1 << LOGW, W = H, PH = 2 * H + 4, PW = 2 * W + 4;
    int idx = blockIdx.x * 256 + threadIdx.x;
    if (idx >= 512 * PH * PW) return;
    int px = idx % PW; int t = idx / PW;
    int py = t % PH;   int bc = t / PH;
    float v = 0.f;
    int r = py - 2, c = px - 2;
    if (r >= 0 && c >= 0 && !(r & 1) && !(c & 1) && (r >> 1) < H && (c >> 1) < W)
        v = -x[((long)bc << (2 * LOGW)) + ((r >> 1) << LOGW) + (c >> 1)];
    dp[idx] = (_Float16)v;
}

// 5x5 path: bilinear 2x upsample of log(|x|+1) written with halo 2:
// (2H+4)x(2W+4) per map (jax.image.resize half-pixel centers, edge-clamped).
template<int LOGW>
__global__ __launch_bounds__(256)
void upbil_log_pad(const float* __restrict__ x, _Float16* __restrict__ mp)
{
    constexpr int H = 1 << LOGW, W = H, PH = 2 * H + 4, PW = 2 * W + 4;
    int idx = blockIdx.x * 256 + threadIdx.x;
    if (idx >= 512 * PH * PW) return;
    int px = idx % PW; int t = idx / PW;
    int py = t % PH;   int bc = t / PH;
    float v = 0.f;
    int oy = py - 2, ox = px - 2;
    if (oy >= 0 && oy < 2 * H && ox >= 0 && ox < 2 * W) {
        int y0, y1; float wy0, wy1;
        if (oy & 1) { y0 = oy >> 1; y1 = y0 + 1; wy0 = 0.75f; wy1 = 0.25f; }
        else        { y1 = oy >> 1; y0 = y1 - 1; wy0 = 0.25f; wy1 = 0.75f; }
        y0 = max(0, y0); y1 = min(H - 1, y1);
        int x0, x1; float wx0, wx1;
        if (ox & 1) { x0 = ox >> 1; x1 = x0 + 1; wx0 = 0.75f; wx1 = 0.25f; }
        else        { x1 = ox >> 1; x0 = x1 - 1; wx0 = 0.25f; wx1 = 0.75f; }
        x0 = max(0, x0); x1 = min(W - 1, x1);
        const float* xb = x + ((long)bc << (2 * LOGW));
        float f00 = logf(fabsf(xb[(y0 << LOGW) + x0]) + 1.0f);
        float f01 = logf(fabsf(xb[(y0 << LOGW) + x1]) + 1.0f);
        float f10 = logf(fabsf(xb[(y1 << LOGW) + x0]) + 1.0f);
        float f11 = logf(fabsf(xb[(y1 << LOGW) + x1]) + 1.0f);
        v = wy0 * (wx0 * f00 + wx1 * f01) + wy1 * (wx0 * f10 + wx1 * f11);
    }
    mp[idx] = (_Float16)v;
}

// ---------------------------------------------------------------------------
// Implicit-GEMM dynamic conv on the WMMA pipe (f16 in, f32 acc), fully
// compile-time geometry (templated on stage LOGW):
//   MODE 0: 3x3 s1 p1 on padded input,         out HxW,   KT=72  (3 WMMA)
//   MODE 1: 4x4 conv on dilated/negated input, out 2Hx2W, KT=128 (4 WMMA)
//   MODE 2: 5x5 s1 p2 on padded upsampled map, out 2Hx2W, KT=200 (7 WMMA)
// One wave = one 16-pixel tile x 8 output channels (M padded to 16).
// Inner loop per element: select-between-two-literal-offsets (lane bit 4),
// one add, one 2-byte load.  No masks, no clamps, no cvt; K-tail is covered
// by the zero-padded A fragment.
// ---------------------------------------------------------------------------
template<int MODE, int LOGW>
__global__ __launch_bounds__(256)
void dynconv_wmma(const _Float16* __restrict__ xp, const _Float16* __restrict__ afrag,
                  const float* __restrict__ bias, float* __restrict__ out,
                  unsigned int* __restrict__ minbuf)
{
    constexpr int KT    = (MODE == 0) ? 72 : (MODE == 1) ? 128 : 200;
    constexpr int NSTEP = (MODE == 0) ? 3  : (MODE == 1) ? 4   : 7;
    constexpr int H  = 1 << LOGW, W = H;
    constexpr int OH = (MODE == 0) ? H : 2 * H;
    constexpr int OW = OH;
    constexpr int PW = (MODE == 0) ? (W + 2) : (2 * W + 4);          // padded row stride
    constexpr int MS = (MODE == 0) ? ((H + 2) * (W + 2))
                                   : ((2 * H + 4) * (2 * W + 4));    // padded map size
    constexpr int KH = (MODE == 0) ? 3 : (MODE == 1) ? 4 : 5;        // kernel extent
    constexpr int TPR = OW / 16;
    constexpr int TPM = OH * OW / 16;

    const int lane = threadIdx.x & 31;
    const int n    = lane & 15;
    const int kgrp = lane >> 4;
    const int gwave = blockIdx.x * 8 + (threadIdx.x >> 5);
    const int b   = gwave / TPM;        // TPM is a power-of-two literal
    const int t   = gwave % TPM;
    const int oy  = t / TPR;
    const int ox0 = (t % TPR) * 16;

    const _Float16* bp8 = xp + (long)b * 8 * MS + oy * PW + ox0 + n;
    const v16h*     ap  = (const v16h*)afrag + (long)b * NSTEP * 32 + lane;

    v8f acc = {};
    #pragma unroll
    for (int s = 0; s < NSTEP; ++s) {
        // ---- A fragment: pre-packed, unconditional 32B load ----
        v16h av = ap[s * 32];
        // ---- B fragment: bare 2-byte gathers at literal offsets ----
        v16h bv;
        #pragma unroll
        for (int h = 0; h < 16; ++h) {
            const int kA = s * 32 + h + ((h < 8) ? 0 : 8);      // compile-time
            const int kB = s * 32 + 8 + h + ((h < 8) ? 0 : 8);  // compile-time
            const int kAc = (kA < KT) ? kA : 0; // tail decodes to a dummy
            const int kBc = (kB < KT) ? kB : 0; // (A fragment is 0 there)
            int offA, offB;
            if (MODE == 1) {
                offA = (kAc >> 4) * MS + ((kAc >> 2) & 3) * PW + (kAc & 3);
                offB = (kBc >> 4) * MS + ((kBc >> 2) & 3) * PW + (kBc & 3);
            } else {
                constexpr int KK = KH * KH;
                offA = (kAc / KK) * MS + ((kAc % KK) / KH) * PW + (kAc % KH);
                offB = (kBc / KK) * MS + ((kBc % KK) / KH) * PW + (kBc % KH);
            }
            const int off = kgrp ? offB : offA; // one cndmask of two literals
            bv[h] = bp8[off];
        }
        // (neg_a, A, neg_b, B, c_mod, C, reuse_a, reuse_b)
        acc = __builtin_amdgcn_wmma_f32_16x16x32_f16(false, av, false, bv,
                                                     (short)0, acc, false, false);
    }

    // D rows 0..7 live in lanes 0-15 (rows 8..15 in lanes 16-31 are padding).
    if (lane < 16) {
        const float*  bp = bias + b * 8;
        float*        ob = out + (long)b * 8 * OH * OW;
        unsigned int* mp = minbuf + b * 8;
        #pragma unroll
        for (int r = 0; r < 8; ++r) {
            float v = acc[r] + bp[r];
            ob[((long)r * OH + oy) * OW + ox0 + n] = v;
            atomicMin(&mp[r], __float_as_uint(fabsf(v)));
        }
    }
}

// ---------------------------------------------------------------------------
// dln(v) = sign(v) * log(log(|v| - mapmin + e) + eps); combine the 3 branches.
// ---------------------------------------------------------------------------
__device__ __forceinline__ float dln_f(float v, float mn)
{
    float s = (v > 0.f) ? 1.f : -1.f;
    float a = fabsf(v) - mn;
    return s * logf(logf(a + EF) + EPSF);
}

__global__ __launch_bounds__(256)
void combine_k(const float* __restrict__ pos_raw, const float* __restrict__ neg_raw,
               const float* __restrict__ mul_raw,
               const unsigned int* __restrict__ minpos,
               const unsigned int* __restrict__ minneg,
               const unsigned int* __restrict__ minmul,
               float* __restrict__ out, int logW)
{
    const int logOW = logW + 1;
    long idx   = (long)blockIdx.x * blockDim.x + threadIdx.x;
    long total = (long)64 * 8 << (2 * logOW);
    if (idx >= total) return;
    int ox = (int)idx & ((1 << logOW) - 1);
    int oy = (int)(idx >> logOW) & ((1 << logOW) - 1);
    int bc = (int)(idx >> (2 * logOW));      // b*8 + c

    float pm = __uint_as_float(minpos[bc]);
    float nm = __uint_as_float(minneg[bc]);
    float mm = __uint_as_float(minmul[bc]);
    float p  = pos_raw[((long)bc << (2 * logW)) + ((oy >> 1) << logW) + (ox >> 1)];
    float ng = neg_raw[idx];
    float ml = mul_raw[idx];
    out[idx] = (dln_f(p, pm) + dln_f(ng, nm)) * dln_f(ml, mm);
}

// ---------------------------------------------------------------------------
// Final dynamic 1x1 conv to 3 channels at 256x256 (memory bound, 24 MAC/px).
// ---------------------------------------------------------------------------
__global__ __launch_bounds__(256)
void final1x1(const float* __restrict__ x, const float* __restrict__ wo,
              const float* __restrict__ bo, float* __restrict__ out)
{
    long idx = (long)blockIdx.x * blockDim.x + threadIdx.x;
    if (idx >= 64L * 3 * 256 * 256) return;
    int px  = (int)idx & 65535;          // oy*256 + ox
    int bo3 = (int)(idx >> 16);          // b*3 + o
    int o  = bo3 % 3;
    int b  = bo3 / 3;
    const float* xb = x + (long)b * 8 * 65536;
    float s = bo[bo3];
    #pragma unroll
    for (int i = 0; i < 8; ++i)
        s += wo[b * 24 + o * 8 + i] * xb[(long)i * 65536 + px];
    out[idx] = s;
}

// ---------------------------------------------------------------------------
// Orchestration.
// ---------------------------------------------------------------------------
extern "C" void kernel_launch(void* const* d_in, const int* in_sizes, int n_in,
                              void* d_out, int out_size, void* d_ws, size_t ws_size,
                              hipStream_t stream)
{
    const int*   ids  = (const int*)  d_in[0];
    const float* ctxc = (const float*)d_in[1];
    const float* lat  = (const float*)d_in[2];
    const float* Wr = (const float*)d_in[3],  *Ur = (const float*)d_in[4];
    const float* Vr = (const float*)d_in[5],  *Br = (const float*)d_in[6];
    const float* Wt = (const float*)d_in[7],  *Ut = (const float*)d_in[8];
    const float* Vt = (const float*)d_in[9],  *Bt = (const float*)d_in[10];
    const float* Wm = (const float*)d_in[11], *Um = (const float*)d_in[12];
    const float* Vm = (const float*)d_in[13], *Bm = (const float*)d_in[14];
    const float* Wo = (const float*)d_in[15], *Uo = (const float*)d_in[16];
    const float* Vo = (const float*)d_in[17], *Bo = (const float*)d_in[18];

    // Workspace carve-up (floats; every block is a multiple of 8 floats = 32B).
    const size_t MAPMAX = (size_t)64 * 8 * 256 * 256;
    float* p = (float*)d_ws;
    float* xA      = p; p += MAPMAX;
    float* xB      = p; p += MAPMAX;
    float* pos_raw = p; p += MAPMAX / 4;   // conv3x3 output is at input res (<=128^2)
    float* neg_raw = p; p += MAPMAX;
    float* mul_raw = p; p += MAPMAX;
    float* ctxb    = p; p += 64 * 32;
    float* w_pr    = p; p += 64 * 8 * 72;
    float* w_pt    = p; p += 64 * 8 * 128;
    float* w_pm    = p; p += 64 * 8 * 200;
    float* b_pr    = p; p += 512;
    float* b_pt    = p; p += 512;
    float* b_pm    = p; p += 512;
    float* w_o     = p; p += 64 * 24;
    float* b_o     = p; p += 192;
    unsigned int* mins = (unsigned int*)p; p += 1536;   // 3 x 512
    _Float16* af_pr = (_Float16*)p; p += (64 * 3 * 32 * 16) / 2;
    _Float16* af_pt = (_Float16*)p; p += (64 * 4 * 32 * 16) / 2;
    _Float16* af_pm = (_Float16*)p; p += (64 * 7 * 32 * 16) / 2;
    _Float16* xpad  = (_Float16*)p; p += (size_t)512 * 130 * 130 / 2;  // halo-1 f16
    _Float16* dil   = (_Float16*)p; p += (size_t)512 * 260 * 260 / 2;  // dilated f16
    _Float16* mpad  = (_Float16*)p; p += (size_t)512 * 260 * 260 / 2;  // upsampled f16

    gather_init<<<64, 256, 0, stream>>>(ids, ctxc, lat, ctxb, xA);

    float* xin  = xA;
    float* xout = xB;

#define RUN_STAGE(LW)                                                                      \
    do {                                                                                   \
        constexpr int H_ = 1 << (LW);                                                      \
        constexpr int PT0 = 512 * (H_ + 2) * (H_ + 2);                                     \
        constexpr int PT1 = 512 * (2 * H_ + 4) * (2 * H_ + 4);                             \
        constexpr int WAVES0 = 64 * H_ * H_ / 16;                                          \
        constexpr int WAVES1 = 64 * 4 * H_ * H_ / 16;                                      \
        pad_copy_f16<LW>  <<<(PT0 + 255) / 256, 256, 0, stream>>>(xin, xpad);              \
        dilate_neg_f16<LW><<<(PT1 + 255) / 256, 256, 0, stream>>>(xin, dil);               \
        upbil_log_pad<LW> <<<(PT1 + 255) / 256, 256, 0, stream>>>(xin, mpad);              \
        dynconv_wmma<0, LW><<<WAVES0 / 8, 256, 0, stream>>>(xpad, af_pr, b_pr,             \
                                                            pos_raw, mins);               \
        dynconv_wmma<1, LW><<<WAVES1 / 8, 256, 0, stream>>>(dil,  af_pt, b_pt,             \
                                                            neg_raw, mins + 512);         \
        dynconv_wmma<2, LW><<<WAVES1 / 8, 256, 0, stream>>>(mpad, af_pm, b_pm,             \
                                                            mul_raw, mins + 1024);        \
        combine_k<<<(64 * 8 * 4 * H_ * H_ + 255) / 256, 256, 0, stream>>>(                 \
            pos_raw, neg_raw, mul_raw, mins, mins + 512, mins + 1024, xout, (LW));         \
    } while (0)

    for (int i = 0; i < 4; ++i) {
        // Per-stage dynamic weights + A-fragment packing.
        dyn_weight<8, 72> <<<64, 256, 0, stream>>>(ctxb, Wr + i * 576,  Ur + i * 4096,
                                                   Vr + i * 36864,  Br + i * 256, w_pr, b_pr);
        dyn_weight<8, 128><<<64, 256, 0, stream>>>(ctxb, Wt + i * 1024, Ut + i * 4096,
                                                   Vt + i * 65536,  Bt + i * 256, w_pt, b_pt);
        dyn_weight<8, 200><<<64, 256, 0, stream>>>(ctxb, Wm + i * 1600, Um + i * 4096,
                                                   Vm + i * 102400, Bm + i * 256, w_pm, b_pm);
        pack_afrag<0><<<(64 * 3 * 32 + 255) / 256, 256, 0, stream>>>(w_pr, af_pr);
        pack_afrag<1><<<(64 * 4 * 32 + 255) / 256, 256, 0, stream>>>(w_pt, af_pt);
        pack_afrag<2><<<(64 * 7 * 32 + 255) / 256, 256, 0, stream>>>(w_pm, af_pm);
        init_min<<<6, 256, 0, stream>>>(mins);

        switch (i) {
            case 0: RUN_STAGE(4); break;
            case 1: RUN_STAGE(5); break;
            case 2: RUN_STAGE(6); break;
            default: RUN_STAGE(7); break;
        }
        float* tmp = xin; xin = xout; xout = tmp;
    }
#undef RUN_STAGE

    dyn_weight<3, 8><<<64, 256, 0, stream>>>(ctxb, Wo, Uo, Vo, Bo, w_o, b_o);
    final1x1<<<(int)((64L * 3 * 256 * 256) / 256), 256, 0, stream>>>(xin, w_o, b_o, (float*)d_out);
}
